// HAPZeroNet_34419867910509
// MI455X (gfx1250) — matrix operand order repaired and checked by reading, not compiled, not run on hardware
//
#include <hip/hip_runtime.h>
#include <math.h>

typedef _Float16 h16;
typedef __attribute__((ext_vector_type(8)))  _Float16 v8h;
typedef __attribute__((ext_vector_type(16))) _Float16 v16h;
typedef __attribute__((ext_vector_type(8)))  float    v8f;

#define LDSP 40  // padded LDS row stride in halves (80B -> 16B aligned rows)

enum { EPI_F32 = 0, EPI_F16 = 1, EPI_GELU_F16 = 2, EPI_RES_F32 = 3,
       EPI_T_F16 = 4, EPI_RES_T_F32 = 5, EPI_ACC_F32 = 6 };

__device__ __forceinline__ float gelu_f(float v) {
  return 0.5f * v * (1.0f + erff(v * 0.70710678118654752f));
}

// ---------------------------------------------------------------------------
// Tiled WMMA GEMM: out = A[M,K] * W[N,K]^T (+bias) with fused epilogues.
// Block tile 128x64, 8 waves (4 M x 2 N), each wave computes 32x32:
// 2 A-frags x 2 B-frags -> 4x v_wmma_f32_16x16x32_f16 per 32-wide K step.
// LDS is double buffered: one barrier per K step; next tile's global loads
// are issued before the WMMAs so their latency hides under matrix math.
// ---------------------------------------------------------------------------
template <int EPI>
__global__ __launch_bounds__(256)
void gemm_wmma_kernel(const h16* __restrict__ A, const h16* __restrict__ W,
                      const float* __restrict__ bias,
                      float* __restrict__ outF, h16* __restrict__ outH,
                      const float* __restrict__ res,
                      int M, int N, int K, int lda, int ldw, int ldo, int ldr,
                      long long sA, long long sW, long long sO, long long sR)
{
  __shared__ h16 As[2 * 128 * LDSP];
  __shared__ h16 Bs[2 * 64 * LDSP];
  const int b = blockIdx.z;
  A += (long long)b * sA;
  W += (long long)b * sW;
  const long long oOff = (long long)b * sO;
  const long long rOff = (long long)b * sR;

  const int tid  = threadIdx.x;
  const int wave = tid >> 5;
  const int lane = tid & 31;
  const int m0 = blockIdx.y * 128;
  const int n0 = blockIdx.x * 64;
  const int wm = (wave >> 1) * 32;   // 0,32,64,96
  const int wn = (wave & 1) * 32;    // 0,32
  const int lr = lane & 15;
  const int hi = lane >> 4;

  // cooperative tile loads: A 128x32 (2 threads/row, 16 halves each),
  //                         B  64x32 (4 threads/row,  8 halves each)
  const int rowA = tid >> 1;          // 0..127
  const int segA = (tid & 1) * 16;    // 0,16
  const int rowB = tid >> 2;          // 0..63
  const int segB = (tid & 3) * 8;     // 0,8,16,24

  const int gmL = m0 + rowA;
  const int gnL = n0 + rowB;
  const bool mOk = gmL < M;
  const bool nOk = gnL < N;
  const h16* Aptr = A + (long long)gmL * lda + segA;
  const h16* Wptr = W + (long long)gnL * ldw + segB;

  v8f a00 = {}, a01 = {}, a10 = {}, a11 = {};

  const int nSteps = K >> 5;  // K is always a multiple of 32 (padded)

  // prologue: tile 0 -> buffer 0
  {
    v8h ra0 = {}, ra1 = {}, rb = {};
    if (mOk) { ra0 = *(const v8h*)(Aptr); ra1 = *(const v8h*)(Aptr + 8); }
    if (nOk) rb = *(const v8h*)(Wptr);
    *(v8h*)(&As[rowA * LDSP + segA])     = ra0;
    *(v8h*)(&As[rowA * LDSP + segA + 8]) = ra1;
    *(v8h*)(&Bs[rowB * LDSP + segB])     = rb;
  }
  __syncthreads();

  for (int i = 0; i < nSteps; ++i) {
    const int cb = i & 1;
    // issue global loads for the next tile (overlap with WMMAs below)
    v8h na0 = {}, na1 = {}, nbv = {};
    const bool more = (i + 1) < nSteps;
    if (more) {
      const int ko = (i + 1) << 5;
      if (mOk) { na0 = *(const v8h*)(Aptr + ko); na1 = *(const v8h*)(Aptr + ko + 8); }
      if (nOk) nbv = *(const v8h*)(Wptr + ko);
    }

    const h16* Ab = &As[cb * 128 * LDSP];
    const h16* Bb = &Bs[cb * 64 * LDSP];
    // A fragment: lane 0-15 row M, K 0-7 & 16-23; lane 16-31 K 8-15 & 24-31
    v16h af0, af1;
    #pragma unroll
    for (int t = 0; t < 8; ++t) {
      af0[t]     = Ab[(wm + lr) * LDSP + hi * 8 + t];
      af0[8 + t] = Ab[(wm + lr) * LDSP + hi * 8 + 16 + t];
    }
    #pragma unroll
    for (int t = 0; t < 8; ++t) {
      af1[t]     = Ab[(wm + 16 + lr) * LDSP + hi * 8 + t];
      af1[8 + t] = Ab[(wm + 16 + lr) * LDSP + hi * 8 + 16 + t];
    }
    // B fragment: lane holds column n, K = hi*8 + t (t<8), 16 + hi*8 + (t-8) (t>=8)
    v16h bf0, bf1;
    #pragma unroll
    for (int t = 0; t < 16; ++t) bf0[t] = Bb[(wn + lr) * LDSP + hi * 8 + (t & 7) + ((t >> 3) << 4)];
    #pragma unroll
    for (int t = 0; t < 16; ++t) bf1[t] = Bb[(wn + 16 + lr) * LDSP + hi * 8 + (t & 7) + ((t >> 3) << 4)];

    a00 = __builtin_amdgcn_wmma_f32_16x16x32_f16(false, af0, false, bf0, (short)0, a00, false, false);
    a01 = __builtin_amdgcn_wmma_f32_16x16x32_f16(false, af0, false, bf1, (short)0, a01, false, false);
    a10 = __builtin_amdgcn_wmma_f32_16x16x32_f16(false, af1, false, bf0, (short)0, a10, false, false);
    a11 = __builtin_amdgcn_wmma_f32_16x16x32_f16(false, af1, false, bf1, (short)0, a11, false, false);

    if (more) {
      const int ob = cb ^ 1;
      *(v8h*)(&As[ob * 128 * LDSP + rowA * LDSP + segA])     = na0;
      *(v8h*)(&As[ob * 128 * LDSP + rowA * LDSP + segA + 8]) = na1;
      *(v8h*)(&Bs[ob * 64 * LDSP + rowB * LDSP + segB])      = nbv;
    }
    __syncthreads();
  }

  // C/D layout: VGPR v, lanes 0-15 -> row v, lanes 16-31 -> row 8+v; col = lane&15
  #pragma unroll
  for (int ma = 0; ma < 2; ++ma) {
    #pragma unroll
    for (int v = 0; v < 8; ++v) {
      const int gm = m0 + wm + ma * 16 + hi * 8 + v;
      if (gm >= M) continue;
      #pragma unroll
      for (int nb = 0; nb < 2; ++nb) {
        const int gn = n0 + wn + nb * 16 + lr;
        if (gn >= N) continue;
        float val;
        if (ma == 0) val = (nb == 0) ? a00[v] : a01[v];
        else         val = (nb == 0) ? a10[v] : a11[v];
        if (bias) val += bias[gn];
        if (EPI == EPI_GELU_F16) val = gelu_f(val);
        if (EPI == EPI_F32) {
          outF[oOff + (long long)gm * ldo + gn] = val;
        } else if (EPI == EPI_F16 || EPI == EPI_GELU_F16) {
          outH[oOff + (long long)gm * ldo + gn] = (h16)val;
        } else if (EPI == EPI_RES_F32) {
          outF[oOff + (long long)gm * ldo + gn] = val + res[rOff + (long long)gm * ldr + gn];
        } else if (EPI == EPI_T_F16) {
          outH[oOff + (long long)gn * ldo + gm] = (h16)val;
        } else if (EPI == EPI_RES_T_F32) {
          outF[oOff + (long long)gn * ldo + gm] = val + res[rOff + (long long)gn * ldr + gm];
        } else if (EPI == EPI_ACC_F32) {
          outF[oOff + (long long)gm * ldo + gn] += val;
        }
      }
    }
  }
}

// ---------------------------------------------------------------------------
// LayerNorm over C=768, output f16 (optionally transposed to [b, c, ldT] + p)
// ---------------------------------------------------------------------------
__global__ __launch_bounds__(256)
void ln_kernel(const float* __restrict__ x, const float* __restrict__ gam,
               const float* __restrict__ bet, h16* __restrict__ out,
               int transposed, int ldT, int rowsPerBatch)
{
  __shared__ float s1[256];
  __shared__ float s2[256];
  const int row = blockIdx.x;
  const int tid = threadIdx.x;
  const float* xr = x + (long long)row * 768;
  const float v0 = xr[tid], v1 = xr[tid + 256], v2 = xr[tid + 512];
  s1[tid] = v0 + v1 + v2;
  s2[tid] = v0 * v0 + v1 * v1 + v2 * v2;
  __syncthreads();
  for (int o = 128; o > 0; o >>= 1) {
    if (tid < o) { s1[tid] += s1[tid + o]; s2[tid] += s2[tid + o]; }
    __syncthreads();
  }
  const float mu   = s1[0] * (1.0f / 768.0f);
  const float var  = s2[0] * (1.0f / 768.0f) - mu * mu;
  const float rstd = rsqrtf(var + 1e-5f);
  const float r0 = (v0 - mu) * rstd * gam[tid]       + bet[tid];
  const float r1 = (v1 - mu) * rstd * gam[tid + 256] + bet[tid + 256];
  const float r2 = (v2 - mu) * rstd * gam[tid + 512] + bet[tid + 512];
  if (!transposed) {
    h16* o = out + (long long)row * 768;
    o[tid] = (h16)r0; o[tid + 256] = (h16)r1; o[tid + 512] = (h16)r2;
  } else {
    const int bb = row / rowsPerBatch;
    const int p  = row % rowsPerBatch;
    h16* o = out + ((long long)bb * 768) * ldT + p;
    o[(long long)tid * ldT]         = (h16)r0;
    o[(long long)(tid + 256) * ldT] = (h16)r1;
    o[(long long)(tid + 512) * ldT] = (h16)r2;
  }
}

// ---------------------------------------------------------------------------
// ReLU6 * scale -> rowmax (to d_out) -> softmax -> padded f16 row
// ---------------------------------------------------------------------------
__global__ __launch_bounds__(256)
void softmax_kernel(const float* __restrict__ raw, h16* __restrict__ out,
                    float* __restrict__ maxOut, int L, int ldi, int ldo, float scale)
{
  __shared__ float sh[256];
  const int row = blockIdx.x;
  const int tid = threadIdx.x;
  const bool valid = tid < L;
  float tv = 0.f;
  if (valid) {
    float r = raw[(long long)row * ldi + tid];
    tv = fminf(fmaxf(r, 0.0f), 6.0f) * scale;
  }
  sh[tid] = valid ? tv : -1e30f;
  __syncthreads();
  for (int o = 128; o > 0; o >>= 1) {
    if (tid < o) sh[tid] = fmaxf(sh[tid], sh[tid + o]);
    __syncthreads();
  }
  const float mx = sh[0];
  __syncthreads();
  if (maxOut && tid == 0) maxOut[row] = mx;
  const float e = valid ? expf(tv - mx) : 0.f;
  sh[tid] = e;
  __syncthreads();
  for (int o = 128; o > 0; o >>= 1) {
    if (tid < o) sh[tid] += sh[tid + o];
    __syncthreads();
  }
  const float inv = 1.0f / sh[0];
  if (tid < ldo) out[(long long)row * ldo + tid] = (h16)(e * inv);
}

// ---------------------------------------------------------------------------
// misc helpers
// ---------------------------------------------------------------------------
__global__ void transpose_f32_kernel(const float* __restrict__ in,
                                     float* __restrict__ out, int R, int S)
{
  __shared__ float t[32][33];
  const int b = blockIdx.z;
  const float* ib = in + (long long)b * R * S;
  float* ob = out + (long long)b * R * S;
  const int s0 = blockIdx.x * 32, r0 = blockIdx.y * 32;
  for (int i = threadIdx.y; i < 32; i += 8) {
    int r = r0 + i, s = s0 + threadIdx.x;
    t[i][threadIdx.x] = (r < R && s < S) ? ib[(long long)r * S + s] : 0.f;
  }
  __syncthreads();
  for (int i = threadIdx.y; i < 32; i += 8) {
    int s = s0 + i, r = r0 + threadIdx.x;
    if (s < S && r < R) ob[(long long)s * R + r] = t[threadIdx.x][i];
  }
}

__global__ void add_f32_kernel(const float* __restrict__ a, const float* __restrict__ b,
                               float* __restrict__ c, long long n)
{
  long long i = (long long)blockIdx.x * 256 + threadIdx.x;
  if (i < n) c[i] = a[i] + b[i];
}

__global__ void fill_f16_kernel(h16* __restrict__ p, long long n)
{
  long long i = (long long)blockIdx.x * 256 + threadIdx.x;
  if (i < n) p[i] = (h16)0.f;
}

__global__ void convert_pad_kernel(const float* __restrict__ src, h16* __restrict__ dst,
                                   int N, int K, int Kp)
{
  long long i = (long long)blockIdx.x * 256 + threadIdx.x;
  if (i >= (long long)N * Kp) return;
  int n = (int)(i / Kp), k = (int)(i % Kp);
  dst[i] = (k < K) ? (h16)src[(long long)n * K + k] : (h16)0.f;
}

// ---------------------------------------------------------------------------
extern "C" void kernel_launch(void* const* d_in, const int* in_sizes, int n_in,
                              void* d_out, int out_size, void* d_ws, size_t ws_size,
                              hipStream_t stream) {
  (void)in_sizes; (void)n_in; (void)out_size; (void)ws_size;
  const int Bb = 128, Cc = 768, Pp = 196, NPp = 85, FFN = 3072, EF = 768, TMH = 512;
  const int Ppad = 224, NPpad = 96;

  const float* x_in    = (const float*)d_in[0];
  const float* parts   = (const float*)d_in[1];
  const float* ea_ln_s = (const float*)d_in[2];
  const float* ea_ln_b = (const float*)d_in[3];
  const float* ea_wqkv = (const float*)d_in[4];
  const float* ea_bqkv = (const float*)d_in[5];
  const float* ea_wp   = (const float*)d_in[6];
  const float* ea_bp   = (const float*)d_in[7];
  const float* da_ln_s = (const float*)d_in[8];
  const float* da_ln_b = (const float*)d_in[9];
  const float* da_wqkv = (const float*)d_in[10];
  const float* da_bqkv = (const float*)d_in[11];
  const float* da_wp   = (const float*)d_in[12];
  const float* da_bp   = (const float*)d_in[13];
  const float* ef_ln_s = (const float*)d_in[14];
  const float* ef_ln_b = (const float*)d_in[15];
  const float* ef_w1   = (const float*)d_in[16];
  const float* ef_b1   = (const float*)d_in[17];
  const float* ef_w2   = (const float*)d_in[18];
  const float* ef_b2   = (const float*)d_in[19];
  const float* f1_ln_s = (const float*)d_in[20];
  const float* f1_ln_b = (const float*)d_in[21];
  const float* f1_w1   = (const float*)d_in[22];
  const float* f1_b1   = (const float*)d_in[23];
  const float* f1_w2   = (const float*)d_in[24];
  const float* f1_b2   = (const float*)d_in[25];
  const float* tm_ln_s = (const float*)d_in[26];
  const float* tm_ln_b = (const float*)d_in[27];
  const float* tm_w1   = (const float*)d_in[28];
  const float* tm_b1   = (const float*)d_in[29];
  const float* tm_w2   = (const float*)d_in[30];
  const float* tm_b2   = (const float*)d_in[31];
  const float* tm_w3   = (const float*)d_in[32];
  const float* tm_b3   = (const float*)d_in[33];

  float* outT      = (float*)d_out;                  // [B,C,P]
  float* attn0_out = outT + (size_t)Bb * Cc * Pp;    // [B,NP]
  float* attn1_out = attn0_out + (size_t)Bb * NPp;   // [B,NP]

  // ---- workspace layout (bump allocator; arena reused across phases) ----
  size_t cur = 0;
  auto alloc = [&](size_t bytes) -> char* {
    cur = (cur + 255) & ~(size_t)255;
    char* p = (char*)d_ws + cur;
    cur += bytes;
    return p;
  };
  h16* wEA  = (h16*)alloc((size_t)3 * Cc * Cc * 2);
  h16* wEAp = (h16*)alloc((size_t)Cc * Cc * 2);
  h16* wDA  = (h16*)alloc((size_t)3 * Cc * Cc * 2);
  h16* wDAp = (h16*)alloc((size_t)Cc * Cc * 2);
  h16* wEF1 = (h16*)alloc((size_t)EF * Cc * 2);
  h16* wEF2 = (h16*)alloc((size_t)Cc * EF * 2);
  h16* wF11 = (h16*)alloc((size_t)FFN * Cc * 2);
  h16* wF12 = (h16*)alloc((size_t)Cc * FFN * 2);
  h16* wTM1 = (h16*)alloc((size_t)TMH * Ppad * 2);
  h16* wTM2 = (h16*)alloc((size_t)TMH * TMH * 2);
  h16* wTM3 = (h16*)alloc((size_t)Pp * TMH * 2);
  float* xp    = (float*)alloc((size_t)Bb * Pp * Cc * 4);   // xp, later feats2
  float* feats = (float*)alloc((size_t)Bb * Pp * Cc * 4);   // feats, later feats3
  float* pA = (float*)alloc((size_t)Bb * NPp * Cc * 4);
  float* pB = (float*)alloc((size_t)Bb * NPp * Cc * 4);
  float* pC = (float*)alloc((size_t)Bb * NPp * Cc * 4);
  cur = (cur + 255) & ~(size_t)255;
  const size_t arenaBase = cur;
  // phase A (attention)
  h16* kea  = (h16*)alloc((size_t)Bb * Pp * Cc * 2);
  h16* vTea = (h16*)alloc((size_t)Bb * Cc * Ppad * 2);
  h16* vTda = (h16*)alloc((size_t)Bb * Cc * NPpad * 2);
  h16* lnb  = (h16*)alloc((size_t)Bb * Pp * Cc * 2);
  h16* fA   = (h16*)alloc((size_t)Bb * Pp * Cc * 2);
  h16* fB   = (h16*)alloc((size_t)Bb * Pp * Cc * 2);
  float* attnraw = (float*)alloc((size_t)Bb * Pp * NPp * 4);
  h16* attnsm = (h16*)alloc((size_t)Bb * 19040 * 2);  // max(85*224, 196*96)
  // phase B (tokenmix) reuses arena
  cur = arenaBase;
  h16* lnT = (h16*)alloc((size_t)Bb * Cc * Ppad * 2);
  h16* h1  = (h16*)alloc((size_t)Bb * Cc * TMH * 2);
  h16* h2  = (h16*)alloc((size_t)Bb * Cc * TMH * 2);
  // phase C (f1 mlp) reuses arena
  cur = arenaBase;
  h16* lnb2 = (h16*)alloc((size_t)Bb * Pp * Cc * 2);
  h16* fC   = (h16*)alloc((size_t)Bb * Pp * Cc * 2);

  // ---- helper launchers ----
  auto conv = [&](const float* src, h16* dst, int N, int K, int Kp) {
    long long tot = (long long)N * Kp;
    convert_pad_kernel<<<dim3((unsigned)((tot + 255) / 256)), dim3(256), 0, stream>>>(src, dst, N, K, Kp);
  };
  auto fill0 = [&](h16* p, long long n) {
    fill_f16_kernel<<<dim3((unsigned)((n + 255) / 256)), dim3(256), 0, stream>>>(p, n);
  };
  auto ln = [&](const float* src, const float* g, const float* be, h16* dst, int rows) {
    ln_kernel<<<dim3((unsigned)rows), dim3(256), 0, stream>>>(src, g, be, dst, 0, 0, 1);
  };
  auto gemm = [&](int epi, const h16* A, const h16* W, const float* bias,
                  float* oF, h16* oH, const float* rs,
                  int M, int N, int K, int lda, int ldw, int ldo, int ldr,
                  long long sA_, long long sW_, long long sO_, long long sR_) {
    dim3 g((unsigned)((N + 63) / 64), (unsigned)((M + 127) / 128), (unsigned)Bb);
    dim3 blk(256);
    switch (epi) {
      case EPI_F32:       gemm_wmma_kernel<EPI_F32><<<g, blk, 0, stream>>>(A, W, bias, oF, oH, rs, M, N, K, lda, ldw, ldo, ldr, sA_, sW_, sO_, sR_); break;
      case EPI_F16:       gemm_wmma_kernel<EPI_F16><<<g, blk, 0, stream>>>(A, W, bias, oF, oH, rs, M, N, K, lda, ldw, ldo, ldr, sA_, sW_, sO_, sR_); break;
      case EPI_GELU_F16:  gemm_wmma_kernel<EPI_GELU_F16><<<g, blk, 0, stream>>>(A, W, bias, oF, oH, rs, M, N, K, lda, ldw, ldo, ldr, sA_, sW_, sO_, sR_); break;
      case EPI_RES_F32:   gemm_wmma_kernel<EPI_RES_F32><<<g, blk, 0, stream>>>(A, W, bias, oF, oH, rs, M, N, K, lda, ldw, ldo, ldr, sA_, sW_, sO_, sR_); break;
      case EPI_T_F16:     gemm_wmma_kernel<EPI_T_F16><<<g, blk, 0, stream>>>(A, W, bias, oF, oH, rs, M, N, K, lda, ldw, ldo, ldr, sA_, sW_, sO_, sR_); break;
      case EPI_RES_T_F32: gemm_wmma_kernel<EPI_RES_T_F32><<<g, blk, 0, stream>>>(A, W, bias, oF, oH, rs, M, N, K, lda, ldw, ldo, ldr, sA_, sW_, sO_, sR_); break;
      case EPI_ACC_F32:   gemm_wmma_kernel<EPI_ACC_F32><<<g, blk, 0, stream>>>(A, W, bias, oF, oH, rs, M, N, K, lda, ldw, ldo, ldr, sA_, sW_, sO_, sR_); break;
    }
  };

  const long long sNPC = (long long)NPp * Cc;
  const long long sPC  = (long long)Pp * Cc;
  const float scl = 1.0f / sqrtf(768.0f);

  // ---- weight conversion to f16 (padded K where needed) ----
  conv(ea_wqkv, wEA, 3 * Cc, Cc, Cc);
  conv(ea_wp, wEAp, Cc, Cc, Cc);
  conv(da_wqkv, wDA, 3 * Cc, Cc, Cc);
  conv(da_wp, wDAp, Cc, Cc, Cc);
  conv(ef_w1, wEF1, EF, Cc, Cc);
  conv(ef_w2, wEF2, Cc, EF, EF);
  conv(f1_w1, wF11, FFN, Cc, Cc);
  conv(f1_w2, wF12, Cc, FFN, FFN);
  conv(tm_w1, wTM1, TMH, Pp, Ppad);
  conv(tm_w2, wTM2, TMH, TMH, TMH);
  conv(tm_w3, wTM3, Pp, TMH, TMH);
  fill0(vTea, (long long)Bb * Cc * Ppad);
  fill0(vTda, (long long)Bb * Cc * NPpad);

  // ---- xp = swapaxes(x) ----
  transpose_f32_kernel<<<dim3((Pp + 31) / 32, (Cc + 31) / 32, Bb), dim3(32, 8), 0, stream>>>(x_in, xp, Cc, Pp);

  // ---- EA shared K / V^T projections (same for both EA attention calls) ----
  ln(xp, ea_ln_s + Cc, ea_ln_b + Cc, lnb, Bb * Pp);
  gemm(EPI_F16, lnb, wEA + (size_t)Cc * Cc, ea_bqkv + Cc, nullptr, kea, nullptr,
       Pp, Cc, Cc, Cc, Cc, Cc, 0, sPC, 0, sPC, 0);
  ln(xp, ea_ln_s + 2 * Cc, ea_ln_b + 2 * Cc, lnb, Bb * Pp);
  gemm(EPI_T_F16, lnb, wEA + (size_t)2 * Cc * Cc, ea_bqkv + 2 * Cc, nullptr, vTea, nullptr,
       Pp, Cc, Cc, Cc, Cc, Ppad, 0, sPC, 0, (long long)Cc * Ppad, 0);

  auto attn_ea = [&](const float* qsrc, const float* resi, float* outP, float* mxOut) {
    ln(qsrc, ea_ln_s, ea_ln_b, lnb, Bb * NPp);
    gemm(EPI_F16, lnb, wEA, ea_bqkv, nullptr, fA, nullptr,
         NPp, Cc, Cc, Cc, Cc, Cc, 0, sNPC, 0, sNPC, 0);
    gemm(EPI_F32, fA, kea, nullptr, attnraw, nullptr, nullptr,
         NPp, Pp, Cc, Cc, Cc, Pp, 0, sNPC, sPC, (long long)NPp * Pp, 0);
    softmax_kernel<<<dim3((unsigned)(Bb * NPp)), dim3(256), 0, stream>>>(
        attnraw, attnsm, mxOut, Pp, Pp, Ppad, scl);
    gemm(EPI_F16, attnsm, vTea, nullptr, nullptr, fB, nullptr,
         NPp, Cc, Ppad, Ppad, Ppad, Cc, 0, (long long)NPp * Ppad, (long long)Cc * Ppad, sNPC, 0);
    gemm(EPI_RES_F32, fB, wEAp, ea_bp, outP, nullptr, resi,
         NPp, Cc, Cc, Cc, Cc, Cc, Cc, sNPC, 0, sNPC, sNPC);
  };
  auto mlp_ef = [&](const float* xin, float* outP) {
    ln(xin, ef_ln_s, ef_ln_b, lnb, Bb * NPp);
    gemm(EPI_GELU_F16, lnb, wEF1, ef_b1, nullptr, fA, nullptr,
         NPp, EF, Cc, Cc, Cc, EF, 0, sNPC, 0, (long long)NPp * EF, 0);
    gemm(EPI_RES_F32, fA, wEF2, ef_b2, outP, nullptr, xin,
         NPp, Cc, EF, EF, EF, Cc, Cc, (long long)NPp * EF, 0, sNPC, sNPC);
  };

  // ---- parts path ----
  attn_ea(parts, parts, pA, attn0_out);            // pA = parts1
  mlp_ef(pA, pB);                                  // pB = parts_out
  add_f32_kernel<<<dim3((unsigned)(((long long)Bb * NPp * Cc + 255) / 256)), dim3(256), 0, stream>>>(
      parts, pB, pC, (long long)Bb * NPp * Cc);    // pC = parts_d
  attn_ea(pC, pC, pA, attn1_out);                  // pA = parts1_d
  mlp_ef(pA, pB);                                  // pB = parts_in

  // ---- DA attention: Q from xp, K/V from parts_in ----
  ln(xp, da_ln_s, da_ln_b, lnb, Bb * Pp);
  gemm(EPI_F16, lnb, wDA, da_bqkv, nullptr, fA, nullptr,
       Pp, Cc, Cc, Cc, Cc, Cc, 0, sPC, 0, sPC, 0);                 // fA = q_da
  ln(pB, da_ln_s + Cc, da_ln_b + Cc, lnb, Bb * NPp);
  gemm(EPI_F16, lnb, wDA + (size_t)Cc * Cc, da_bqkv + Cc, nullptr, fB, nullptr,
       NPp, Cc, Cc, Cc, Cc, Cc, 0, sNPC, 0, sNPC, 0);              // fB = k_da
  ln(pB, da_ln_s + 2 * Cc, da_ln_b + 2 * Cc, lnb, Bb * NPp);
  gemm(EPI_T_F16, lnb, wDA + (size_t)2 * Cc * Cc, da_bqkv + 2 * Cc, nullptr, vTda, nullptr,
       NPp, Cc, Cc, Cc, Cc, NPpad, 0, sNPC, 0, (long long)Cc * NPpad, 0);
  gemm(EPI_F32, fA, fB, nullptr, attnraw, nullptr, nullptr,
       Pp, NPp, Cc, Cc, Cc, NPp, 0, sPC, sNPC, (long long)Pp * NPp, 0);
  softmax_kernel<<<dim3((unsigned)(Bb * Pp)), dim3(256), 0, stream>>>(
      attnraw, attnsm, nullptr, NPp, NPp, NPpad, scl);
  gemm(EPI_F16, attnsm, vTda, nullptr, nullptr, fA, nullptr,
       Pp, Cc, NPpad, NPpad, NPpad, Cc, 0, (long long)Pp * NPpad, (long long)Cc * NPpad, sPC, 0);
  gemm(EPI_RES_F32, fA, wDAp, da_bp, feats, nullptr, xp,
       Pp, Cc, Cc, Cc, Cc, Cc, Cc, sPC, 0, sPC, sPC);              // feats = xp + proj

  // ---- tokenmix ----
  fill0(lnT, (long long)Bb * Cc * Ppad);
  ln_kernel<<<dim3((unsigned)(Bb * Pp)), dim3(256), 0, stream>>>(
      feats, tm_ln_s, tm_ln_b, lnT, 1, Ppad, Pp);                  // [B,C,Ppad]
  gemm(EPI_GELU_F16, lnT, wTM1, tm_b1, nullptr, h1, nullptr,
       Cc, TMH, Ppad, Ppad, Ppad, TMH, 0, (long long)Cc * Ppad, 0, (long long)Cc * TMH, 0);
  gemm(EPI_GELU_F16, h1, wTM2, tm_b2, nullptr, h2, nullptr,
       Cc, TMH, TMH, TMH, TMH, TMH, 0, (long long)Cc * TMH, 0, (long long)Cc * TMH, 0);
  gemm(EPI_RES_T_F32, h2, wTM3, tm_b3, xp /*feats2*/, nullptr, feats,
       Cc, Pp, TMH, TMH, TMH, Cc, Cc, (long long)Cc * TMH, 0, sPC, sPC);

  // ---- f1 MLP (hidden 3072 chunked 4x768), feats3 = feats2 + mlp(feats2) ----
  ln(xp, f1_ln_s, f1_ln_b, lnb2, Bb * Pp);
  for (int j = 0; j < 4; ++j) {
    gemm(EPI_GELU_F16, lnb2, wF11 + (size_t)j * 768 * Cc, f1_b1 + j * 768, nullptr, fC, nullptr,
         Pp, 768, Cc, Cc, Cc, 768, 0, sPC, 0, (long long)Pp * 768, 0);
    gemm(j == 0 ? EPI_RES_F32 : EPI_ACC_F32, fC, wF12 + (size_t)j * 768,
         j == 0 ? f1_b2 : nullptr, feats, nullptr, xp,
         Pp, Cc, 768, 768, FFN, Cc, Cc, (long long)Pp * 768, 0, sPC, sPC);
  }

  // ---- final transpose to [B, C, P] ----
  transpose_f32_kernel<<<dim3((Cc + 31) / 32, (Pp + 31) / 32, Bb), dim3(32, 8), 0, stream>>>(
      feats, outT, Pp, Cc);
}
